// FullAttention_43482248905153
// MI455X (gfx1250) — matrix operand order repaired and checked by reading, not compiled, not run on hardware
//
#include <hip/hip_runtime.h>
#include <hip/hip_bf16.h>
#include <math.h>

typedef __attribute__((ext_vector_type(16))) __bf16 v16bf;
typedef __attribute__((ext_vector_type(8)))  __bf16 v8bf;
typedef __attribute__((ext_vector_type(8)))  float  v8f;

#define B_     2
#define L_     2048
#define H_     16
#define E_     64
#define QTILE  128
#define WAVES  8
#define KSTRIP 32

// padded LDS row strides; all piece offsets stay 16B aligned.
#define KLD 72   // K rows (bf16): 64 + 8  (144 B stride)
#define VLD 40   // Vt rows (bf16):32 + 8  ( 80 B stride)
#define SLD 36   // S rows (f32):  32 + 4  (144 B stride)

static __device__ inline v16bf cat8(v8bf a, v8bf b) {
  return __builtin_shufflevector(a, b, 0,1,2,3,4,5,6,7,8,9,10,11,12,13,14,15);
}

static __device__ inline v8bf cvt8(float4 a, float4 b, float s) {
  v8bf r;
  r[0] = (__bf16)(a.x * s); r[1] = (__bf16)(a.y * s);
  r[2] = (__bf16)(a.z * s); r[3] = (__bf16)(a.w * s);
  r[4] = (__bf16)(b.x * s); r[5] = (__bf16)(b.y * s);
  r[6] = (__bf16)(b.z * s); r[7] = (__bf16)(b.w * s);
  return r;
}

// ---- CDNA5 async global->LDS copy (ASYNCcnt-tracked), 16B per lane ----
static __device__ inline void async_copy_b128(void* lds, const void* g) {
  unsigned ldsa = (unsigned)(uintptr_t)lds;          // low 32 bits = LDS offset
  unsigned long long ga = (unsigned long long)(uintptr_t)g;
  asm volatile("global_load_async_to_lds_b128 %0, %1, off"
               :: "v"(ldsa), "v"(ga) : "memory");
}
static __device__ inline void async_wait_le2() {
  asm volatile("s_wait_asynccnt 0x2" ::: "memory");
}
static __device__ inline void async_wait_0() {
  asm volatile("s_wait_asynccnt 0x0" ::: "memory");
}

// ================= pre-pass: K -> bf16 [b][h][key][e] =================
__global__ __launch_bounds__(256) void cvt_k(const float* __restrict__ K,
                                             __bf16* __restrict__ Kb) {
  size_t f = ((size_t)blockIdx.x * 256 + threadIdx.x) * 8;   // dst flat index
  int e0  = (int)(f % E_);
  int key = (int)((f / E_) % L_);
  int h   = (int)((f / ((size_t)E_ * L_)) % H_);
  int b   = (int)( f / ((size_t)E_ * L_ * H_));
  const float* src = K + (((size_t)b * L_ + key) * H_ + h) * E_ + e0;
  float4 a = *(const float4*)src;
  float4 c = *(const float4*)(src + 4);
  *(v8bf*)(Kb + f) = cvt8(a, c, 1.0f);
}

// ============ pre-pass: V -> bf16 transposed [b][h][e][key] ============
__global__ __launch_bounds__(256) void cvt_vt(const float* __restrict__ V,
                                              __bf16* __restrict__ Vt) {
  __shared__ __align__(16) __bf16 T[E_][VLD];
  const int nstr  = L_ / KSTRIP;
  const int strip = blockIdx.x % nstr;
  const int bh    = blockIdx.x / nstr;          // b*H + h
  const int j32   = strip * KSTRIP;
  const int tid   = threadIdx.x;

  const float* src = V + (((size_t)(bh / H_) * L_) * H_ + (bh % H_)) * E_;
  {
    int key = tid >> 3, e0 = (tid & 7) * 8;
    const float* vp = src + (size_t)(j32 + key) * (H_ * E_) + e0;
    float4 a = *(const float4*)vp;
    float4 c = *(const float4*)(vp + 4);
    T[e0 + 0][key] = (__bf16)a.x;  T[e0 + 1][key] = (__bf16)a.y;
    T[e0 + 2][key] = (__bf16)a.z;  T[e0 + 3][key] = (__bf16)a.w;
    T[e0 + 4][key] = (__bf16)c.x;  T[e0 + 5][key] = (__bf16)c.y;
    T[e0 + 6][key] = (__bf16)c.z;  T[e0 + 7][key] = (__bf16)c.w;
  }
  __syncthreads();
  {
    int e = tid >> 2, k0 = (tid & 3) * 8;
    v8bf row = *(const v8bf*)&T[e][k0];
    *(v8bf*)(Vt + ((size_t)bh * E_ + e) * L_ + j32 + k0) = row;
  }
}

// ======================= shared compute core =======================
template <typename KLOAD>
static __device__ inline void fa_core(
    const float* Qb, float* Og, int b, int h, int qrow0, int nstrips,
    __bf16 (*KlBuf)[KSTRIP][KLD], __bf16 (*VtBuf)[E_][VLD],
    float (*Sl)[16][SLD], KLOAD stage, bool dbuf)
{
  const int tid  = threadIdx.x;
  const int w    = tid >> 5;
  const int lane = tid & 31;
  const int hi   = lane >> 4;
  const int ln   = lane & 15;
  const int strideL = H_ * E_;

  // Q A-fragments, prescaled by sqrt(E)=8 (reference multiplies scores by 8)
  v16bf qf[2];
#pragma unroll
  for (int c = 0; c < 2; ++c) {
    const float* qr = Qb + (size_t)(qrow0 + ln) * strideL + c * 32 + hi * 8;
    float4 a0 = *(const float4*)(qr +  0);
    float4 a1 = *(const float4*)(qr +  4);
    float4 b0 = *(const float4*)(qr + 16);
    float4 b1 = *(const float4*)(qr + 20);
    qf[c] = cat8(cvt8(a0, a1, 8.0f), cvt8(b0, b1, 8.0f));
  }

  v8f acc[4];
#pragma unroll
  for (int cE = 0; cE < 4; ++cE) acc[cE] = (v8f){0.f,0.f,0.f,0.f,0.f,0.f,0.f,0.f};
  float m_row = -__builtin_inff();
  float l_row = 0.f;

  for (int j = 0; j < nstrips; ++j) {
    const int j32 = j * KSTRIP;
    stage(j);                       // produce/retire strip j, prime strip j+1
    __syncthreads();

    const int cur = dbuf ? (j & 1) : 0;
    __bf16 (*Kl)[KLD] = KlBuf[cur];
    __bf16 (*Vt)[VLD] = VtBuf[cur];

    if (j32 <= qrow0 + 15) {        // wave-uniform causal skip
      v8f s0 = (v8f){0.f,0.f,0.f,0.f,0.f,0.f,0.f,0.f};
      v8f s1 = (v8f){0.f,0.f,0.f,0.f,0.f,0.f,0.f,0.f};
#pragma unroll
      for (int c = 0; c < 2; ++c) {
        const int eb = c * 32 + hi * 8;
        v16bf kf0 = cat8(*(const v8bf*)&Kl[ln     ][eb],
                         *(const v8bf*)&Kl[ln     ][eb + 16]);
        v16bf kf1 = cat8(*(const v8bf*)&Kl[ln + 16][eb],
                         *(const v8bf*)&Kl[ln + 16][eb + 16]);
        s0 = __builtin_amdgcn_wmma_f32_16x16x32_bf16(false, qf[c], false, kf0,
                                                     (short)0, s0, false, false);
        s1 = __builtin_amdgcn_wmma_f32_16x16x32_bf16(false, qf[c], false, kf1,
                                                     (short)0, s1, false, false);
      }

      // C-layout -> A-layout transpose of raw scores through LDS
#pragma unroll
      for (int r = 0; r < 8; ++r) {
        Sl[w][r + 8 * hi][ln]      = s0[r];
        Sl[w][r + 8 * hi][ln + 16] = s1[r];
      }
      const float* srow = &Sl[w][ln][0];
      float4 y0 = *(const float4*)(srow + hi * 8);
      float4 y1 = *(const float4*)(srow + hi * 8 + 4);
      float4 y2 = *(const float4*)(srow + 16 + hi * 8);
      float4 y3 = *(const float4*)(srow + 20 + hi * 8);
      float x[16] = { y0.x, y0.y, y0.z, y0.w,  y1.x, y1.y, y1.z, y1.w,
                      y2.x, y2.y, y2.z, y2.w,  y3.x, y3.y, y3.z, y3.w };

      if (j32 + KSTRIP - 1 > qrow0) {       // diagonal strips only
        const int qi = qrow0 + ln;
#pragma unroll
        for (int i = 0; i < 16; ++i) {
          const int k = (i >> 3) * 16 + hi * 8 + (i & 7);
          x[i] = (j32 + k <= qi) ? x[i] : -__builtin_inff();
        }
      }

      float mx01 = fmaxf(fmaxf(x[0], x[1]),   fmaxf(x[2], x[3]));
      float mx23 = fmaxf(fmaxf(x[4], x[5]),   fmaxf(x[6], x[7]));
      float mx45 = fmaxf(fmaxf(x[8], x[9]),   fmaxf(x[10], x[11]));
      float mx67 = fmaxf(fmaxf(x[12], x[13]), fmaxf(x[14], x[15]));
      float mx = fmaxf(fmaxf(mx01, mx23), fmaxf(mx45, mx67));
      mx = fmaxf(mx, __shfl_xor(mx, 16, 32));

      const float mn    = fmaxf(m_row, mx);
      const float alpha = __expf(m_row - mn);
      m_row = mn;

      float p[16];
      float sum = 0.f;
#pragma unroll
      for (int i = 0; i < 16; ++i) { p[i] = __expf(x[i] - mn); sum += p[i]; }
      sum += __shfl_xor(sum, 16, 32);
      l_row = l_row * alpha + sum;

      v16bf pf;
#pragma unroll
      for (int i = 0; i < 16; ++i) pf[i] = (__bf16)p[i];

#pragma unroll
      for (int r = 0; r < 8; ++r) {
        const float aC = __shfl(alpha, r + 8 * hi, 32);
#pragma unroll
        for (int cE = 0; cE < 4; ++cE) acc[cE][r] *= aC;
      }

#pragma unroll
      for (int cE = 0; cE < 4; ++cE) {
        v16bf vf = cat8(*(const v8bf*)&Vt[cE * 16 + ln][hi * 8],
                        *(const v8bf*)&Vt[cE * 16 + ln][hi * 8 + 16]);
        acc[cE] = __builtin_amdgcn_wmma_f32_16x16x32_bf16(false, pf, false, vf,
                                                          (short)0, acc[cE], false, false);
      }
    }
    __syncthreads();
  }

  float linv[8];
#pragma unroll
  for (int r = 0; r < 8; ++r)
    linv[r] = 1.0f / __shfl(l_row, r + 8 * hi, 32);
#pragma unroll
  for (int r = 0; r < 8; ++r) {
    const int qi = qrow0 + r + 8 * hi;
    float* op = Og + ((size_t)(b * L_ + qi) * H_ + h) * E_ + ln;
#pragma unroll
    for (int cE = 0; cE < 4; ++cE)
      op[cE * 16] = acc[cE][r] * linv[r];
  }
}

// ============ async main kernel: double-buffered ASYNCcnt staging ============
__global__ __launch_bounds__(256) void fa_fwd_causal_async(
    const float* __restrict__ Qg, const __bf16* __restrict__ Kb,
    const __bf16* __restrict__ Vtg, float* __restrict__ Og)
{
  __shared__ __align__(16) __bf16 Kl2[2][KSTRIP][KLD];
  __shared__ __align__(16) __bf16 Vt2[2][E_][VLD];
  __shared__ __align__(16) float  Sl[WAVES][16][SLD];

  const int ntiles = L_ / QTILE;
  const int qtile  = blockIdx.x % ntiles;
  const int h      = (blockIdx.x / ntiles) % H_;
  const int b      = blockIdx.x / (ntiles * H_);
  const int tid    = threadIdx.x;
  const int qrow0  = qtile * QTILE + (tid >> 5) * 16;
  const int nstrips = (qtile * QTILE + QTILE) / KSTRIP;

  const float*  Qb  = Qg  + ((size_t)b * L_ * H_ + h) * E_;
  const __bf16* KbBH  = Kb  + ((size_t)(b * H_ + h)) * L_ * E_;   // [key][e]
  const __bf16* VtBH  = Vtg + ((size_t)(b * H_ + h)) * E_ * L_;   // [e][key]

  // per-thread async chunk coordinates (one K chunk + one Vt chunk per strip)
  const int krow = tid >> 3, kch = (tid & 7) * 8;   // K: 32 rows x 8 chunks
  const int vrow = tid >> 2, vch = (tid & 3) * 8;   // Vt: 64 rows x 4 chunks

  auto issue = [&](int js, int buf) {
    async_copy_b128(&Kl2[buf][krow][kch],
                    KbBH + ((size_t)(js * KSTRIP) + krow) * E_ + kch);
    async_copy_b128(&Vt2[buf][vrow][vch],
                    VtBH + (size_t)vrow * L_ + js * KSTRIP + vch);
  };

  issue(0, 0);                                   // prologue: strip 0 in flight
  auto stage = [&](int j) {
    if (j + 1 < nstrips) { issue(j + 1, (j + 1) & 1); async_wait_le2(); }
    else                 { async_wait_0(); }
  };

  fa_core(Qb, Og, b, h, qrow0, nstrips, Kl2, Vt2, Sl, stage, /*dbuf=*/true);
}

// ============ synchronous fallback (no workspace needed) ============
__global__ __launch_bounds__(256) void fa_fwd_causal_sync(
    const float* __restrict__ Qg, const float* __restrict__ Kg,
    const float* __restrict__ Vg, float* __restrict__ Og)
{
  __shared__ __align__(16) __bf16 Kl1[1][KSTRIP][KLD];
  __shared__ __align__(16) __bf16 Vt1[1][E_][VLD];
  __shared__ __align__(16) float  Sl[WAVES][16][SLD];

  const int ntiles = L_ / QTILE;
  const int qtile  = blockIdx.x % ntiles;
  const int h      = (blockIdx.x / ntiles) % H_;
  const int b      = blockIdx.x / (ntiles * H_);
  const int tid    = threadIdx.x;
  const int qrow0  = qtile * QTILE + (tid >> 5) * 16;
  const int nstrips = (qtile * QTILE + QTILE) / KSTRIP;
  const int strideL = H_ * E_;

  const float* Qb = Qg + ((size_t)b * L_ * H_ + h) * E_;
  const float* Kb = Kg + ((size_t)b * L_ * H_ + h) * E_;
  const float* Vb = Vg + ((size_t)b * L_ * H_ + h) * E_;

  const int skey = tid >> 3, se = (tid & 7) * 8;
  auto stage = [&](int j) {
    const int j32 = j * KSTRIP;
    const float* kp = Kb + (size_t)(j32 + skey) * strideL + se;
    float4 ka = *(const float4*)kp;
    float4 kb = *(const float4*)(kp + 4);
    *(v8bf*)&Kl1[0][skey][se] = cvt8(ka, kb, 1.0f);
    const float* vp = Vb + (size_t)(j32 + skey) * strideL + se;
    float4 va = *(const float4*)vp;
    float4 vb = *(const float4*)(vp + 4);
    Vt1[0][se + 0][skey] = (__bf16)va.x;  Vt1[0][se + 1][skey] = (__bf16)va.y;
    Vt1[0][se + 2][skey] = (__bf16)va.z;  Vt1[0][se + 3][skey] = (__bf16)va.w;
    Vt1[0][se + 4][skey] = (__bf16)vb.x;  Vt1[0][se + 5][skey] = (__bf16)vb.y;
    Vt1[0][se + 6][skey] = (__bf16)vb.z;  Vt1[0][se + 7][skey] = (__bf16)vb.w;
  };

  fa_core(Qb, Og, b, h, qrow0, nstrips, Kl1, Vt1, Sl, stage, /*dbuf=*/false);
}

extern "C" void kernel_launch(void* const* d_in, const int* in_sizes, int n_in,
                              void* d_out, int out_size, void* d_ws, size_t ws_size,
                              hipStream_t stream) {
  (void)in_sizes; (void)n_in; (void)out_size;
  const float* q = (const float*)d_in[0];
  const float* k = (const float*)d_in[1];
  const float* v = (const float*)d_in[2];
  // d_in[3] is the causal mask; structurally triu(k=1), baked into the kernels.
  float* out = (float*)d_out;
  dim3 grid(B_ * H_ * (L_ / QTILE));

  const size_t half = (size_t)B_ * H_ * L_ * E_ * sizeof(__bf16);
  if (ws_size >= 2 * half) {
    __bf16* kb  = (__bf16*)d_ws;
    __bf16* vtg = (__bf16*)((char*)d_ws + half);
    cvt_k <<<B_ * H_ * L_ * E_ / (256 * 8), 256, 0, stream>>>(k, kb);
    cvt_vt<<<B_ * H_ * (L_ / KSTRIP),       256, 0, stream>>>(v, vtg);
    fa_fwd_causal_async<<<grid, 256, 0, stream>>>(q, kb, vtg, out);
  } else {
    fa_fwd_causal_sync<<<grid, 256, 0, stream>>>(q, k, v, out);
  }
}
